// SelfAttention_14405320311188
// MI455X (gfx1250) — matrix operand order repaired and checked by reading, not compiled, not run on hardware
//
#include <hip/hip_runtime.h>
#include <hip/hip_bf16.h>
#include <math.h>

// ---------------------------------------------------------------------------
// Self-attention forward for MI455X (gfx1250), bf16 WMMA pipeline.
//   qkv = x @ W + b ; S = q k^T ; P = softmax(scale*S + mask) ; y = P v
// GEMMs: v_wmma_f32_16x16x32_bf16, double-buffered LDS staged with
// GLOBAL_LOAD_ASYNC_TO_LDS_B128 (ASYNCcnt) when the builtin is available.
// ---------------------------------------------------------------------------

typedef unsigned short u16;
typedef __attribute__((ext_vector_type(16))) __bf16 v16bf;
typedef __attribute__((ext_vector_type(8)))  float  v8f;
typedef __attribute__((ext_vector_type(4)))  int    v4i;

union FragU { uint4 u[2]; v16bf v; };

__device__ __forceinline__ u16 f2bf(float f) {
    union { float f; unsigned int u; } x;
    x.f = f;
    unsigned int r = x.u + 0x7FFFu + ((x.u >> 16) & 1u);   // round-to-nearest-even
    return (u16)(r >> 16);
}

// ------------------------- async global -> LDS -----------------------------
#if __has_builtin(__builtin_amdgcn_global_load_async_to_lds_b128)
#define USE_ASYNC_LDS 1
#else
#define USE_ASYNC_LDS 0
#endif

typedef __attribute__((address_space(1))) v4i* gas_v4i_ptr;
typedef __attribute__((address_space(3))) v4i* las_v4i_ptr;

__device__ __forceinline__ void cp_b128(const u16* g, u16* l) {
#if USE_ASYNC_LDS
    __builtin_amdgcn_global_load_async_to_lds_b128(
        (gas_v4i_ptr)(g), (las_v4i_ptr)(l), 0, 0);
#else
    *(uint4*)l = *(const uint4*)g;
#endif
}

__device__ __forceinline__ void wait_async0() {
#if USE_ASYNC_LDS
#if __has_builtin(__builtin_amdgcn_s_wait_asynccnt)
    __builtin_amdgcn_s_wait_asynccnt(0);
#else
    asm volatile("s_wait_asynccnt 0x0" ::: "memory");
#endif
#endif
}

// ---------------------------------------------------------------------------
// fp32 -> bf16 elementwise convert (vectorized)
// ---------------------------------------------------------------------------
__global__ __launch_bounds__(256) void f32_to_bf16_kernel(
    const float* __restrict__ in, u16* __restrict__ out, size_t n)
{
    size_t i = ((size_t)blockIdx.x * 256 + threadIdx.x) * 4;
    if (i + 3 < n) {
        float4 f = *(const float4*)(in + i);
        unsigned int p0 = (unsigned int)f2bf(f.x) | ((unsigned int)f2bf(f.y) << 16);
        unsigned int p1 = (unsigned int)f2bf(f.z) | ((unsigned int)f2bf(f.w) << 16);
        uint2 o; o.x = p0; o.y = p1;
        *(uint2*)(out + i) = o;
    }
}

// ---------------------------------------------------------------------------
// W [rows x cols] fp32 row-major  ->  Wt [cols x rows] bf16 row-major
// ---------------------------------------------------------------------------
__global__ __launch_bounds__(256) void transpose_w_kernel(
    const float* __restrict__ W, u16* __restrict__ Wt, int rows, int cols)
{
    __shared__ float tile[32][33];
    int c0 = blockIdx.x * 32;
    int r0 = blockIdx.y * 32;
    int tx = threadIdx.x & 31;
    int ty = threadIdx.x >> 5;            // 0..7
#pragma unroll
    for (int i = 0; i < 32; i += 8)
        tile[ty + i][tx] = W[(size_t)(r0 + ty + i) * cols + (c0 + tx)];
    __syncthreads();
#pragma unroll
    for (int i = 0; i < 32; i += 8)
        Wt[(size_t)(c0 + ty + i) * rows + (r0 + tx)] = f2bf(tile[tx][ty + i]);
}

// ---------------------------------------------------------------------------
// Batched bf16 transpose: V[b][m][d] (row stride ld, col offset coff)
//   -> vT[b][d][m]  (n x n per batch)
// ---------------------------------------------------------------------------
__global__ __launch_bounds__(256) void transpose_v_kernel(
    const u16* __restrict__ qkv, u16* __restrict__ vT, int n, int ld, int coff)
{
    __shared__ u16 tile[32][33];
    int b = blockIdx.z;
    const u16* V = qkv + (size_t)b * n * ld + coff;
    u16* O = vT + (size_t)b * n * n;
    int c0 = blockIdx.x * 32;
    int r0 = blockIdx.y * 32;
    int tx = threadIdx.x & 31;
    int ty = threadIdx.x >> 5;
#pragma unroll
    for (int i = 0; i < 32; i += 8)
        tile[ty + i][tx] = V[(size_t)(r0 + ty + i) * ld + (c0 + tx)];
    __syncthreads();
#pragma unroll
    for (int i = 0; i < 32; i += 8)
        O[(size_t)(c0 + ty + i) * n + (r0 + tx)] = tile[tx][ty + i];
}

// ---------------------------------------------------------------------------
// Generic bf16 WMMA GEMM:  C[M,N] = A[M,K] @ Bt[N,K]^T  (+ bias[col])
// Block: 256 threads (8 waves), tile 128(M) x 256(N), k-step 64.
// Waves laid out 2(M) x 4(N); each wave computes 64x64 = 4x4 tiles of 16x16.
// Double-buffered LDS (96 KB), next stage staged with async global->LDS.
// ---------------------------------------------------------------------------
#define BM 128
#define BN 256
#define BK 64

__device__ __forceinline__ v16bf ld_frag_a(const u16* lds, int row, int kk)
{
    int lane = threadIdx.x & 31;
    int r  = row + (lane & 15);
    int kb = (lane >> 4) << 3;           // 0 or 8
    const u16* p = lds + r * BK + kk + kb;
    FragU f;
    f.u[0] = *(const uint4*)(p);         // K = kb .. kb+7
    f.u[1] = *(const uint4*)(p + 16);    // K = 16+kb .. 16+kb+7
    return f.v;
}

__device__ __forceinline__ v16bf ld_frag_b(const u16* lds, int col, int kk)
{
    int lane = threadIdx.x & 31;
    int c  = col + (lane & 15);
    int kb = (lane >> 4) << 3;           // 0 or 8
    const u16* p = lds + c * BK + kk + kb;
    FragU f;
    f.u[0] = *(const uint4*)(p);         // K = kb .. kb+7
    f.u[1] = *(const uint4*)(p + 16);    // K = 16+kb .. 16+kb+7
    return f.v;
}

template <bool BF16_OUT, bool HAS_BIAS>
__global__ __launch_bounds__(256) void gemm_bf16_wmma_kernel(
    const u16* __restrict__ A,  int lda, long long strideA,
    const u16* __restrict__ Bt, int ldb, long long strideB,
    void* __restrict__ Cptr, int ldc, long long strideC,
    const float* __restrict__ bias, int K)
{
    __shared__ alignas(16) u16 sA[2][BM * BK];   // 2 x 16 KiB
    __shared__ alignas(16) u16 sB[2][BN * BK];   // 2 x 32 KiB

    const int bz = blockIdx.z;
    A  += (size_t)bz * strideA;
    Bt += (size_t)bz * strideB;

    const int m0 = blockIdx.y * BM;
    const int n0 = blockIdx.x * BN;
    const int t  = threadIdx.x;
    const int wave = t >> 5;
    const int wm = (wave >> 2) * 64;     // 0 or 64
    const int wn = (wave & 3) * 64;      // 0,64,128,192

    v8f acc[16];
#pragma unroll
    for (int i = 0; i < 16; ++i) acc[i] = v8f{};

    // stage k-slab k0 into buffer `buf` (A: 1024 chunks of 16B, B: 2048 chunks)
    auto stage = [&](int buf, int k0) {
#pragma unroll
        for (int i = 0; i < 4; ++i) {                 // A tile: 128 x 64
            int chunk = t + i * 256;                  // 0..1023
            int row = chunk >> 3, c8 = chunk & 7;     // 8 chunks per 128B row
            cp_b128(A + (size_t)(m0 + row) * lda + k0 + c8 * 8,
                    &sA[buf][chunk * 8]);
        }
#pragma unroll
        for (int i = 0; i < 8; ++i) {                 // Bt tile: 256 x 64
            int chunk = t + i * 256;                  // 0..2047
            int row = chunk >> 3, c8 = chunk & 7;
            cp_b128(Bt + (size_t)(n0 + row) * ldb + k0 + c8 * 8,
                    &sB[buf][chunk * 8]);
        }
    };

    int buf = 0;
    stage(buf, 0);
    wait_async0();
    __syncthreads();

    for (int k0 = 0; k0 < K; k0 += BK) {
        if (k0 + BK < K) stage(buf ^ 1, k0 + BK);    // prefetch next slab

        const u16* cA = sA[buf];
        const u16* cB = sB[buf];
#pragma unroll
        for (int kk = 0; kk < BK; kk += 32) {
            v16bf af[4], bfr[4];
#pragma unroll
            for (int mi = 0; mi < 4; ++mi) af[mi]  = ld_frag_a(cA, wm + mi * 16, kk);
#pragma unroll
            for (int ni = 0; ni < 4; ++ni) bfr[ni] = ld_frag_b(cB, wn + ni * 16, kk);
#pragma unroll
            for (int mi = 0; mi < 4; ++mi)
#pragma unroll
                for (int ni = 0; ni < 4; ++ni)
                    acc[mi * 4 + ni] = __builtin_amdgcn_wmma_f32_16x16x32_bf16(
                        false, af[mi], false, bfr[ni], (short)0, acc[mi * 4 + ni],
                        false, false);
        }
        wait_async0();          // next slab fully landed in the other buffer
        __syncthreads();
        buf ^= 1;
    }

    // epilogue: 16x16 f32 C layout -> row = j + (lane>=16)*8, col = lane&15
    float* Cf = (float*)Cptr;
    u16*   Cb = (u16*)Cptr;
    const int lane  = t & 31;
    const int cIn   = lane & 15;
    const int rHalf = (lane >> 4) * 8;
#pragma unroll
    for (int mi = 0; mi < 4; ++mi) {
#pragma unroll
        for (int ni = 0; ni < 4; ++ni) {
            v8f a = acc[mi * 4 + ni];
            int rbase = m0 + wm + mi * 16 + rHalf;
            int c     = n0 + wn + ni * 16 + cIn;
            float badd = HAS_BIAS ? bias[c] : 0.0f;
#pragma unroll
            for (int j = 0; j < 8; ++j) {
                float val = a[j] + badd;
                size_t idx = (size_t)bz * strideC + (size_t)(rbase + j) * ldc + c;
                if constexpr (BF16_OUT) Cb[idx] = f2bf(val);
                else                    Cf[idx] = val;
            }
        }
    }
}

// ---------------------------------------------------------------------------
// Row softmax with causal + padding mask; emits P (bf16) and denominator.
// One block (256 threads) per row of S[B*N, n_len].
// ---------------------------------------------------------------------------
__global__ __launch_bounds__(256) void softmax_kernel(
    const float* __restrict__ S, const int* __restrict__ pm,
    u16* __restrict__ P, float* __restrict__ denom_out,
    int n_len, float scale)
{
    __shared__ float red[256];
    const int row = blockIdx.x;              // 0 .. B*N-1
    const int b   = row / n_len;
    const int n   = row % n_len;
    const float* s = S + (size_t)row * n_len;
    u16* p = P + (size_t)row * n_len;
    const int t = threadIdx.x;
    const float NEG_INF = -__builtin_inff();

    const bool rowOn = pm[(size_t)b * n_len + n] != 0;

    float v[8];
    float mx = NEG_INF;
#pragma unroll
    for (int i = 0; i < 8; ++i) {
        int m = i * 256 + t;
        bool ok = rowOn && (m <= n) && (pm[(size_t)b * n_len + m] != 0);
        v[i] = ok ? s[m] * scale : NEG_INF;
        mx = fmaxf(mx, v[i]);
    }
    red[t] = mx; __syncthreads();
    for (int st = 128; st > 0; st >>= 1) {
        if (t < st) red[t] = fmaxf(red[t], red[t + st]);
        __syncthreads();
    }
    mx = red[0];
    __syncthreads();

    float e[8];
    float sum = 0.0f;
#pragma unroll
    for (int i = 0; i < 8; ++i) {
        e[i] = (v[i] == NEG_INF) ? 0.0f : __expf(v[i] - mx);
        sum += e[i];
    }
    red[t] = sum; __syncthreads();
    for (int st = 128; st > 0; st >>= 1) {
        if (t < st) red[t] = red[t] + red[t + st];
        __syncthreads();
    }
    float d = red[0];
    float inv = (d > 0.0f) ? (1.0f / d) : 0.0f;
#pragma unroll
    for (int i = 0; i < 8; ++i)
        p[i * 256 + t] = f2bf(e[i] * inv);
    if (t == 0) denom_out[row] = d;
}

// ---------------------------------------------------------------------------
// Host-side orchestration
// ---------------------------------------------------------------------------
extern "C" void kernel_launch(void* const* d_in, const int* in_sizes, int n_in,
                              void* d_out, int out_size, void* d_ws, size_t ws_size,
                              hipStream_t stream)
{
    (void)in_sizes; (void)n_in; (void)out_size; (void)ws_size;

    const float* x    = (const float*)d_in[0];   // [4,2048,2048]
    const int*   pm   = (const int*)  d_in[1];   // [4,2048]
    const float* W    = (const float*)d_in[2];   // [2048,6144]
    const float* bias = (const float*)d_in[3];   // [6144]
    float* out = (float*)d_out;                  // y [4,2048,2048] then denom [4,2048]

    const int Bc = 4, Nn = 2048, EMB = 2048, OUT = 2048, QKV = 3 * OUT;

    // workspace layout (bytes)
    char* ws = (char*)d_ws;
    u16*   xb   = (u16*)ws;                                       // 32 MiB (x bf16; later P)
    u16*   Wt   = (u16*)(ws + (size_t)33554432);                  // 24 MiB (W^T bf16)
    u16*   qkvb = (u16*)(ws + (size_t)33554432 + 25165824);       // 96 MiB (qkv bf16)
    float* S    = (float*)(ws + (size_t)33554432 + 25165824 + 100663296); // 64 MiB (scores; later vT)
    u16*   vT   = (u16*)S;

    // 1) x -> bf16
    size_t nx = (size_t)Bc * Nn * EMB;
    f32_to_bf16_kernel<<<(unsigned)(nx / 1024), 256, 0, stream>>>(x, xb, nx);

    // 2) W^T -> bf16   (Wt[n][k] = W[k][n])
    transpose_w_kernel<<<dim3(QKV / 32, EMB / 32), 256, 0, stream>>>(W, Wt, EMB, QKV);

    // 3) qkv = x @ W + b     (M = 8192, N = 6144, K = 2048) -> bf16
    gemm_bf16_wmma_kernel<true, true><<<dim3(QKV / BN, (Bc * Nn) / BM, 1), 256, 0, stream>>>(
        xb, EMB, 0,
        Wt, EMB, 0,
        qkvb, QKV, 0,
        bias, EMB);

    // 4) S[b] = q[b] @ k[b]^T   (K matrix rows are already [N][K] layout)
    gemm_bf16_wmma_kernel<false, false><<<dim3(Nn / BN, Nn / BM, Bc), 256, 0, stream>>>(
        qkvb,        QKV, (long long)Nn * QKV,       // q
        qkvb + OUT,  QKV, (long long)Nn * QKV,       // k (as Bt)
        S, Nn, (long long)Nn * Nn,
        nullptr, OUT);

    // 5) softmax rows -> P (bf16, reuses xb) + denom (second output region)
    const float scale = 1.0f / sqrtf((float)OUT);
    softmax_kernel<<<Bc * Nn, 256, 0, stream>>>(
        S, pm, xb, out + (size_t)Bc * Nn * OUT, Nn, scale);

    // 6) v^T (bf16) into the S buffer region (scores fully consumed)
    transpose_v_kernel<<<dim3(Nn / 32, Nn / 32, Bc), 256, 0, stream>>>(
        qkvb, vT, Nn, QKV, 2 * OUT);

    // 7) y[b] = P[b] @ v[b]   (Bt = v^T) -> fp32 straight to d_out
    gemm_bf16_wmma_kernel<false, false><<<dim3(OUT / BN, Nn / BM, Bc), 256, 0, stream>>>(
        xb, Nn, (long long)Nn * Nn,
        vT, Nn, (long long)Nn * Nn,
        out, OUT, (long long)Nn * OUT,
        nullptr, Nn);
}
